// VisionMultiStageSNN_62989990363218
// MI455X (gfx1250) — compile-verified, matrix operand order checked by reading, not run on hardware
//
#include <hip/hip_runtime.h>
#include <hip/hip_bf16.h>

typedef __attribute__((ext_vector_type(16))) _Float16 v16h;
typedef __attribute__((ext_vector_type(8)))  float    v8f;

#define DECAY   0.95f
#define KPANEL  8             // k-blocks (of 32) per LDS panel for stage 1
#define NPANEL  12            // 96 k-blocks total (N_IN=3072)

#if __has_builtin(__builtin_amdgcn_tensor_load_to_lds) && __has_builtin(__builtin_amdgcn_s_wait_tensorcnt)
#define HAVE_TDM 1
#else
#define HAVE_TDM 0
#endif

// ---------------------------------------------------------------------------
// Prep: convert fp32 matrices (R rows, I cols, row-major) into f16 fragment
// order [row16][kb][lane][16], matching the 16x16x32 f16 WMMA operand layout
// (lane<16: row=lane, K {0..7,16..23}; lane>=16: row=lane-16, K {8..15,24..31}).
// Used for the weights (B operands) AND for the spike tensor (A operands,
// R = T*B = 8192 rows), so the hot kernel only ever does 32B fragment loads
// or pure linear DMA copies.
// ---------------------------------------------------------------------------
__global__ __launch_bounds__(256) void convert_w(const float* __restrict__ W,
                                                 _Float16* __restrict__ Wf,
                                                 int R, int I) {
  int c = blockIdx.x * 256 + threadIdx.x;     // one 8-element chunk per thread
  int cpr = I >> 3;                           // chunks per row
  if (c >= R * cpr) return;
  int g  = c / cpr;
  int k0 = (c - g * cpr) << 3;
  const float* src = W + (size_t)g * I + k0;
  float4 a0 = *(const float4*)src;
  float4 a1 = *(const float4*)(src + 4);
  int kb = k0 >> 5;
  int kk = k0 & 31;                           // 0, 8, 16, 24
  int lane = (g & 15) + ((kk & 8) ? 16 : 0);
  int p0   = (kk & 16) ? 8 : 0;
  int kbc  = I >> 5;
  _Float16* d = Wf + (((size_t)(g >> 4) * kbc + kb) << 9) + lane * 16 + p0;
  d[0]=(_Float16)a0.x; d[1]=(_Float16)a0.y; d[2]=(_Float16)a0.z; d[3]=(_Float16)a0.w;
  d[4]=(_Float16)a1.x; d[5]=(_Float16)a1.y; d[6]=(_Float16)a1.z; d[7]=(_Float16)a1.w;
}

#if HAVE_TDM
typedef __attribute__((ext_vector_type(4))) unsigned int v4u;
typedef __attribute__((ext_vector_type(8))) int v8i;
typedef __attribute__((ext_vector_type(4))) int v4i;

// Linear TDM copy: ndw DWORDs from global gaddr -> LDS lds_addr.
// D# packed per CDNA5 ISA 8.3/8.4: group0 {count=1, lds_addr, global_addr,
// type=2}; group1 {data_size=4B, tensor_dim0=ndw, tensor_dim1=1,
// tile_dim0=ndw, tile_dim1=1, strides=ndw}; groups 2/3 zero (<=2D tensor).
__device__ __forceinline__ void tdm_copy(uint32_t lds_addr, uint64_t gaddr,
                                         uint32_t ndw) {
  v4u g0;
  g0.x = 1u;                                                   // count=1
  g0.y = lds_addr;
  g0.z = (uint32_t)gaddr;
  g0.w = ((uint32_t)(gaddr >> 32) & 0x01FFFFFFu) | (2u << 30); // ga[56:32]|type=2
  v8i g1;
  g1[0] = (int)(2u << 16);                        // data_size = 4 bytes
  g1[1] = (int)((ndw & 0xFFFFu) << 16);           // tensor_dim0[15:0]
  g1[2] = (int)(((ndw >> 16) & 0xFFFFu) | (1u << 16)); // td0[31:16]|td1[15:0]=1
  g1[3] = (int)((ndw & 0xFFFFu) << 16);           // td1[31:16]=0 | tile_dim0
  g1[4] = 1;                                      // tile_dim1=1, tile_dim2=0
  g1[5] = (int)ndw;                               // td0_stride[31:0]
  g1[6] = (int)((ndw & 0xFFFFu) << 16);           // td0s[47:32]=0|td1s[15:0]
  g1[7] = (int)(ndw >> 16);                       // td1_stride[47:16]
  v4i gz = (v4i){0, 0, 0, 0};
#if defined(__clang_major__) && (__clang_major__ >= 23)
  v8i gz8 = (v8i){0, 0, 0, 0, 0, 0, 0, 0};
  __builtin_amdgcn_tensor_load_to_lds(g0, g1, gz, gz, gz8, 0);
#else
  __builtin_amdgcn_tensor_load_to_lds(g0, g1, gz, gz, 0);
#endif
}

__device__ __forceinline__ uint32_t lds_off(const void* p) {
  return (uint32_t)(unsigned long long)p;
}
#endif

// Build an f16 A-fragment from an f32 row-major LDS buffer (stage 2/3 inputs).
__device__ __forceinline__ v16h cvt_frag_f32(const float* __restrict__ buf,
                                             int stride, int kb, int lane) {
  int m = lane & 15;
  const float* r = buf + m * stride + (kb << 5) + ((lane >> 4) << 3);
  float4 q0 = *(const float4*)(r);
  float4 q1 = *(const float4*)(r + 4);
  float4 q2 = *(const float4*)(r + 16);
  float4 q3 = *(const float4*)(r + 20);
  v16h a;
  a[0]=(_Float16)q0.x;  a[1]=(_Float16)q0.y;  a[2]=(_Float16)q0.z;  a[3]=(_Float16)q0.w;
  a[4]=(_Float16)q1.x;  a[5]=(_Float16)q1.y;  a[6]=(_Float16)q1.z;  a[7]=(_Float16)q1.w;
  a[8]=(_Float16)q2.x;  a[9]=(_Float16)q2.y;  a[10]=(_Float16)q2.z; a[11]=(_Float16)q2.w;
  a[12]=(_Float16)q3.x; a[13]=(_Float16)q3.y; a[14]=(_Float16)q3.z; a[15]=(_Float16)q3.w;
  return a;
}

// One LIF-GEMM stage for one wave: acc = DECAY*v + A@B; spike; ds_add combine.
template<int NTH, int KB, int OSTRIDE>
__device__ __forceinline__ void run_stage(const float* __restrict__ cin, int instride,
                                          float* __restrict__ cout,
                                          const _Float16* __restrict__ Wf,
                                          int row16base, float* __restrict__ vbase,
                                          int ncol0, float gate, int lane) {
  v8f acc[NTH];
#pragma unroll
  for (int nt = 0; nt < NTH; ++nt) {
    const float4* vp = (const float4*)(vbase + ((nt * 32 + lane) << 3));
    float4 lo = vp[0], hi = vp[1];
    acc[nt][0]=lo.x*DECAY; acc[nt][1]=lo.y*DECAY; acc[nt][2]=lo.z*DECAY; acc[nt][3]=lo.w*DECAY;
    acc[nt][4]=hi.x*DECAY; acc[nt][5]=hi.y*DECAY; acc[nt][6]=hi.z*DECAY; acc[nt][7]=hi.w*DECAY;
  }
#pragma unroll 2
  for (int kb = 0; kb < KB; ++kb) {
    v16h a = cvt_frag_f32(cin, instride, kb, lane);
    const _Float16* wp = Wf + (((size_t)row16base * KB + kb) << 9) + lane * 16;
#pragma unroll
    for (int nt = 0; nt < NTH; ++nt) {
      v16h b = *(const v16h*)(wp + (((size_t)nt * KB) << 9));
      acc[nt] = __builtin_amdgcn_wmma_f32_16x16x32_f16(false, a, false, b,
                                                       (short)0, acc[nt], false, false);
    }
  }
  int m0 = (lane >> 4) << 3;
  int nn = lane & 15;
#pragma unroll
  for (int nt = 0; nt < NTH; ++nt) {
#pragma unroll
    for (int r = 0; r < 8; ++r) {
      float s = (acc[nt][r] >= 1.0f) ? 1.0f : 0.0f;
      acc[nt][r] -= s;
      atomicAdd(&cout[(m0 + r) * OSTRIDE + ncol0 + nt * 16 + nn], s * gate);
    }
    float4* vp = (float4*)(vbase + ((nt * 32 + lane) << 3));
    vp[0] = make_float4(acc[nt][0], acc[nt][1], acc[nt][2], acc[nt][3]);
    vp[1] = make_float4(acc[nt][4], acc[nt][5], acc[nt][6], acc[nt][7]);
  }
}

// ---------------------------------------------------------------------------
// Persistent main kernel: one workgroup per 16-row batch tile; 16 wave32s,
// wave w = (expert w>>1, output-half w&1).  All T steps, all 3 stages.
// Stage-1 A data arrives via double-buffered TDM DMA into LDS.
// ---------------------------------------------------------------------------
__global__ __launch_bounds__(512) void snn_moe_lif_persistent(
    const _Float16* __restrict__ Xf,
    const _Float16* __restrict__ W1f, const _Float16* __restrict__ W2f,
    const _Float16* __restrict__ W3f,
    const float* __restrict__ g1, const float* __restrict__ g2,
    const float* __restrict__ g3,
    float* __restrict__ V1, float* __restrict__ V2, float* __restrict__ V3,
    float* __restrict__ O1, float* __restrict__ O2, float* __restrict__ O3,
    float* __restrict__ out) {
#if HAVE_TDM
  __shared__ __align__(1024) _Float16 xsh0[KPANEL * 32 * 16];  // 8 KB
  __shared__ __align__(1024) _Float16 xsh1[KPANEL * 32 * 16];  // 8 KB
#endif
  __shared__ float c1s[16 * 256];                              // 16 KB
  __shared__ float c2s[16 * 256];                              // 16 KB
  __shared__ float c3s[16 * 128];                              // 8 KB

  const int wg = blockIdx.x;
  const int tid = threadIdx.x;
  const int lane = tid & 31;
  const int w = tid >> 5;
  const int e = w >> 1;
  const int half = w & 1;
  const int wgB = wg * 16;

  float* V1wg = V1 + (size_t)wg * 32768;
  float* V2wg = V2 + (size_t)wg * 32768;
  float* V3wg = V3 + (size_t)wg * 16384;
  float* O1wg = O1 + (size_t)wg * 4096;
  float* O2wg = O2 + (size_t)wg * 4096;
  float* O3wg = O3 + (size_t)wg * 2048;

  // deterministic per-launch init of this tile's state
  for (int i = tid; i < 32768; i += 512) { V1wg[i] = 0.f; V2wg[i] = 0.f; }
  for (int i = tid; i < 16384; i += 512) V3wg[i] = 0.f;
  for (int i = tid; i < 4096; i += 512) { O1wg[i] = 0.f; O2wg[i] = 0.f; }
  for (int i = tid; i < 2048; i += 512) O3wg[i] = 0.f;

  const float ge1 = g1[e], ge2 = g2[e], ge3 = g3[e];
  float* v1base = V1wg + ((e * 16 + half * 8) << 8);
  float* v2base = V2wg + ((e * 16 + half * 8) << 8);
  float* v3base = V3wg + ((e * 8 + half * 4) << 8);

  for (int t = 0; t < 16; ++t) {
    __syncthreads();  // previous step's consumers of c buffers are done
    // x slab for this (t, tile): 96 kb-fragments, contiguous, 96 KB
    const _Float16* xslab = Xf + (size_t)(t * 32 + wg) * 96 * 512;
#if HAVE_TDM
    if (w == 0)  // kick DMA of panel 0 immediately; overlaps zeroing + v init
      tdm_copy(lds_off(xsh0), (uint64_t)(unsigned long long)xslab, 2048);
#endif
    for (int i = tid; i < 4096; i += 512) { c1s[i] = 0.f; c2s[i] = 0.f; }
    for (int i = tid; i < 2048; i += 512) c3s[i] = 0.f;

    // ---------------- Stage 1: h1 = x_t @ W1[e]^T, K = 3072 ----------------
    v8f acc[8];
#pragma unroll
    for (int nt = 0; nt < 8; ++nt) {
      const float4* vp = (const float4*)(v1base + ((nt * 32 + lane) << 3));
      float4 lo = vp[0], hi = vp[1];
      acc[nt][0]=lo.x*DECAY; acc[nt][1]=lo.y*DECAY; acc[nt][2]=lo.z*DECAY; acc[nt][3]=lo.w*DECAY;
      acc[nt][4]=hi.x*DECAY; acc[nt][5]=hi.y*DECAY; acc[nt][6]=hi.z*DECAY; acc[nt][7]=hi.w*DECAY;
    }
    for (int p = 0; p < NPANEL; ++p) {
#if HAVE_TDM
      if (w == 0 && p + 1 < NPANEL)  // DMA next panel into the other buffer
        tdm_copy(lds_off((p & 1) ? xsh0 : xsh1),
                 (uint64_t)(unsigned long long)xslab + (uint64_t)(p + 1) * 8192,
                 2048);
      if (p + 1 < NPANEL) __builtin_amdgcn_s_wait_tensorcnt(1);
      else                __builtin_amdgcn_s_wait_tensorcnt(0);
      __syncthreads();  // panel p resident; all waves may read
      const _Float16* cur = (p & 1) ? xsh1 : xsh0;
#endif
#pragma unroll 2
      for (int kbl = 0; kbl < KPANEL; ++kbl) {
        int kb = p * KPANEL + kbl;
#if HAVE_TDM
        v16h a = *(const v16h*)(cur + ((kbl * 32 + lane) << 4));
#else
        v16h a = *(const v16h*)(xslab + ((size_t)kb << 9) + lane * 16);
#endif
        const _Float16* wp =
            W1f + (((size_t)(e * 16 + half * 8) * 96 + kb) << 9) + lane * 16;
        __builtin_prefetch(wp + 512, 0, 3);  // next k-block weight fragment
#pragma unroll
        for (int nt = 0; nt < 8; ++nt) {
          v16h b = *(const v16h*)(wp + (((size_t)nt * 96) << 9));
          acc[nt] = __builtin_amdgcn_wmma_f32_16x16x32_f16(false, a, false, b,
                                                           (short)0, acc[nt], false, false);
        }
      }
#if HAVE_TDM
      __syncthreads();  // all reads of this buffer done before DMA reuses it
#endif
    }
    {  // LIF + gated combine into c1
      int m0 = (lane >> 4) << 3, nn = lane & 15, nc0 = half * 128;
#pragma unroll
      for (int nt = 0; nt < 8; ++nt) {
#pragma unroll
        for (int r = 0; r < 8; ++r) {
          float s = (acc[nt][r] >= 1.0f) ? 1.0f : 0.0f;
          acc[nt][r] -= s;
          atomicAdd(&c1s[(m0 + r) * 256 + nc0 + nt * 16 + nn], s * ge1);
        }
        float4* vp = (float4*)(v1base + ((nt * 32 + lane) << 3));
        vp[0] = make_float4(acc[nt][0], acc[nt][1], acc[nt][2], acc[nt][3]);
        vp[1] = make_float4(acc[nt][4], acc[nt][5], acc[nt][6], acc[nt][7]);
      }
    }
    __syncthreads();  // c1 complete
    for (int i = tid; i < 4096; i += 512) O1wg[i] += c1s[i];

    // ---------------- Stage 2: h2 = c1 @ W2[e]^T, K = 256 ------------------
    run_stage<8, 8, 256>(c1s, 256, c2s, W2f, e * 16 + half * 8, v2base,
                         half * 128, ge2, lane);
    __syncthreads();  // c2 complete
    for (int i = tid; i < 4096; i += 512) O2wg[i] += c2s[i];

    // ---------------- Stage 3: h3 = c2 @ W3[e]^T, K = 256 ------------------
    run_stage<4, 8, 128>(c2s, 256, c3s, W3f, e * 8 + half * 4, v3base,
                         half * 64, ge3, lane);
    __syncthreads();  // c3 complete
    for (int i = tid; i < 2048; i += 512) O3wg[i] += c3s[i];
  }

  const float inv = 1.0f / 16.0f;
  for (int i = tid; i < 4096; i += 512) {
    int m = i >> 8, n = i & 255;
    out[(wgB + m) * 256 + n] = O1wg[i] * inv;
    out[131072 + (wgB + m) * 256 + n] = O2wg[i] * inv;
  }
  for (int i = tid; i < 2048; i += 512) {
    int m = i >> 7, n = i & 127;
    out[262144 + (wgB + m) * 128 + n] = O3wg[i] * inv;
  }
}

extern "C" void kernel_launch(void* const* d_in, const int* in_sizes, int n_in,
                              void* d_out, int out_size, void* d_ws, size_t ws_size,
                              hipStream_t stream) {
  const float* spikes = (const float*)d_in[0];
  const float* W1 = (const float*)d_in[1];
  const float* W2 = (const float*)d_in[2];
  const float* W3 = (const float*)d_in[3];
  const float* g1 = (const float*)d_in[4];
  const float* g2 = (const float*)d_in[5];
  const float* g3 = (const float*)d_in[6];

  char* ws = (char*)d_ws;
  _Float16* W1f = (_Float16*)ws; ws += (size_t)6291456 * 2;   // 8*256*3072 f16
  _Float16* W2f = (_Float16*)ws; ws += (size_t)524288 * 2;    // 8*256*256
  _Float16* W3f = (_Float16*)ws; ws += (size_t)262144 * 2;    // 8*128*256
  _Float16* Xf  = (_Float16*)ws; ws += (size_t)25165824 * 2;  // 16*512*3072 f16
  float* V1 = (float*)ws; ws += (size_t)1048576 * 4;          // 32 wg * 32768
  float* V2 = (float*)ws; ws += (size_t)1048576 * 4;
  float* V3 = (float*)ws; ws += (size_t)524288 * 4;
  float* O1 = (float*)ws; ws += (size_t)131072 * 4;
  float* O2 = (float*)ws; ws += (size_t)131072 * 4;
  float* O3 = (float*)ws; ws += (size_t)65536 * 4;

  convert_w<<<(2048 * 384 + 255) / 256, 256, 0, stream>>>(W1, W1f, 2048, 3072);
  convert_w<<<(2048 * 32 + 255) / 256, 256, 0, stream>>>(W2, W2f, 2048, 256);
  convert_w<<<(1024 * 32 + 255) / 256, 256, 0, stream>>>(W3, W3f, 1024, 256);
  convert_w<<<(8192 * 384 + 255) / 256, 256, 0, stream>>>(spikes, Xf, 8192, 3072);

  snn_moe_lif_persistent<<<32, 512, 0, stream>>>(Xf, W1f, W2f, W3f,
                                                 g1, g2, g3, V1, V2, V3,
                                                 O1, O2, O3, (float*)d_out);
}
